// MyModel_27831388078362
// MI455X (gfx1250) — matrix operand order repaired and checked by reading, not compiled, not run on hardware
//
#include <hip/hip_runtime.h>
#include <hip/hip_bf16.h>

// ---------------------------------------------------------------------------
// MI455X (gfx1250) implementation of the GraphModel reference.
// All matmuls run through V_WMMA_F32_16X16X32_BF16 (wave32).  Each wave
// computes a 32x32 output tile (2x2 register blocking -> 4 WMMA per 8 b128
// fragment loads, 16 FLOP/byte of operand traffic).  Activations/weights are
// staged as bf16 with K padded to a multiple of 32; weights are stored
// transposed ([N,Kp] row-major) so both A and B WMMA fragments are two
// contiguous 128-bit loads per lane, in contiguous k order (k = 16*hi + s)
// used identically for A and B.  BN statistics, softmax, ELU and the loss
// stay in f32.
// ---------------------------------------------------------------------------

typedef __bf16 v16bf __attribute__((ext_vector_type(16)));
typedef float  v8f   __attribute__((ext_vector_type(8)));

using bf16 = __hip_bfloat16;

__device__ __forceinline__ bf16 f2bf(float x) { return __float2bfloat16(x); }

union FragU { uint4 u[2]; v16bf v; };
static_assert(sizeof(FragU) == 32, "fragment must be 32 bytes");

#define WMMA_BF16(a, b, c) \
    __builtin_amdgcn_wmma_f32_16x16x32_bf16(false, (a), false, (b), (short)0, (c), false, false)

// ---------------------------------------------------------------------------
// Generic bf16 GEMM: C[M,N] = A[M,Kp] * Bt[N,Kp]^T (+bias), f32 accumulate.
// Block = 256 threads = 8 waves arranged 4(M) x 2(N), 32x32 tile per wave:
// block tile 128x64.  Requires: M % 128 == 0, N % 64 == 0, Kp % 32 == 0,
// pointers 16B aligned.
// ---------------------------------------------------------------------------
__global__ void __launch_bounds__(256)
k_gemm_bf16(const bf16* __restrict__ A, const bf16* __restrict__ Bt,
            float* __restrict__ C, const float* __restrict__ bias,
            int M, int N, int Kp)
{
    const int lane = threadIdx.x & 31;
    const int wave = threadIdx.x >> 5;
    const int lo   = lane & 15;
    const int hi   = lane >> 4;
    const int m0 = blockIdx.y * 128 + (wave & 3) * 32;
    const int n0 = blockIdx.x * 64  + (wave >> 2) * 32;

    const bf16* arow0 = A  + (size_t)(m0 + lo) * Kp;
    const bf16* arow1 = arow0 + (size_t)16 * Kp;
    const bf16* brow0 = Bt + (size_t)(n0 + lo) * Kp;
    const bf16* brow1 = brow0 + (size_t)16 * Kp;
    const int abase = hi * 16;   // A frag: lanes 0-15 hold K k..k+15; lanes 16-31 hold k+16..k+31
    const int bbase = hi * 16;   // B frag: lanes 0-15 hold K k..k+15; lanes 16-31 hold k+16..k+31

    v8f c00 = {0.f,0.f,0.f,0.f,0.f,0.f,0.f,0.f};
    v8f c01 = c00, c10 = c00, c11 = c00;

    for (int k0 = 0; k0 < Kp; k0 += 32) {
        FragU a0, a1, b0, b1;
        const uint4* ap0 = (const uint4*)(arow0 + k0 + abase);
        a0.u[0] = ap0[0]; a0.u[1] = ap0[1];
        const uint4* ap1 = (const uint4*)(arow1 + k0 + abase);
        a1.u[0] = ap1[0]; a1.u[1] = ap1[1];
        const uint4* bp0 = (const uint4*)(brow0 + k0 + bbase);
        b0.u[0] = bp0[0]; b0.u[1] = bp0[1];
        const uint4* bp1 = (const uint4*)(brow1 + k0 + bbase);
        b1.u[0] = bp1[0]; b1.u[1] = bp1[1];
        __builtin_prefetch(arow0 + k0 + 128, 0, 1);
        __builtin_prefetch(arow1 + k0 + 128, 0, 1);
        __builtin_prefetch(brow0 + k0 + 128, 0, 1);
        __builtin_prefetch(brow1 + k0 + 128, 0, 1);
        c00 = WMMA_BF16(a0.v, b0.v, c00);
        c01 = WMMA_BF16(a0.v, b1.v, c01);
        c10 = WMMA_BF16(a1.v, b0.v, c10);
        c11 = WMMA_BF16(a1.v, b1.v, c11);
    }

    const int col0 = n0 + lo;
    const int col1 = col0 + 16;
    const float bv0 = bias ? bias[col0] : 0.0f;
    const float bv1 = bias ? bias[col1] : 0.0f;
    // C frag: VGPR r -> row (tileRow + hi*8 + r), column = tileCol + lo.
    float* cr0 = C + (size_t)(m0 + hi * 8) * N;
    float* cr1 = C + (size_t)(m0 + 16 + hi * 8) * N;
#pragma unroll
    for (int r = 0; r < 8; ++r) {
        cr0[(size_t)r * N + col0] = c00[r] + bv0;
        cr0[(size_t)r * N + col1] = c01[r] + bv1;
        cr1[(size_t)r * N + col0] = c10[r] + bv0;
        cr1[(size_t)r * N + col1] = c11[r] + bv1;
    }
}

// ---------------------------------------------------------------------------
// f32 [M,K] -> bf16 [M,Kp] with zero padding of columns K..Kp.
// ---------------------------------------------------------------------------
__global__ void k_cast_pad(const float* __restrict__ src, bf16* __restrict__ dst,
                           int M, int K, int Kp)
{
    size_t idx = (size_t)blockIdx.x * blockDim.x + threadIdx.x;
    if (idx >= (size_t)M * Kp) return;
    int    k = (int)(idx % Kp);
    size_t i = idx / Kp;
    dst[idx] = (k < K) ? f2bf(src[i * (size_t)K + k]) : f2bf(0.f);
}

// ---------------------------------------------------------------------------
// Weight f32 [K,N] (row-major) -> bf16 transposed [N,Kp] with zero K-pad.
// ---------------------------------------------------------------------------
__global__ void k_cast_transpose(const float* __restrict__ src, bf16* __restrict__ dst,
                                 int K, int N, int Kp)
{
    size_t idx = (size_t)blockIdx.x * blockDim.x + threadIdx.x;
    if (idx >= (size_t)N * Kp) return;
    int k = (int)(idx % Kp);
    int n = (int)(idx / Kp);
    dst[idx] = (k < K) ? f2bf(src[(size_t)k * N + n]) : f2bf(0.f);
}

// ---------------------------------------------------------------------------
// BatchNorm (training mode) statistics: one block per feature column.
// ---------------------------------------------------------------------------
__global__ void k_bn_stats(const float* __restrict__ X, int M, int F,
                           float* __restrict__ mean, float* __restrict__ rstd)
{
    const int j = blockIdx.x;
    float s = 0.f, ss = 0.f;
    for (int i = threadIdx.x; i < M; i += blockDim.x) {
        float v = X[(size_t)i * F + j];
        s += v; ss += v * v;
    }
    __shared__ float sh[256], sh2[256];
    sh[threadIdx.x] = s; sh2[threadIdx.x] = ss;
    __syncthreads();
    for (int st = 128; st > 0; st >>= 1) {
        if (threadIdx.x < st) { sh[threadIdx.x] += sh[threadIdx.x + st];
                                sh2[threadIdx.x] += sh2[threadIdx.x + st]; }
        __syncthreads();
    }
    if (threadIdx.x == 0) {
        float m = sh[0] / M;
        float v = sh2[0] / M - m * m;
        mean[j] = m;
        rstd[j] = rsqrtf(v + 1e-5f);
    }
}

// BN + ReLU, emit bf16 into dst[i*ldo + colOff + j], zero-padding j in [F,Fp).
__global__ void k_bn_relu_bf16(const float* __restrict__ X,
                               const float* __restrict__ mean, const float* __restrict__ rstd,
                               const float* __restrict__ g, const float* __restrict__ be,
                               bf16* __restrict__ dst, int M, int F, int ldo, int colOff, int Fp)
{
    size_t idx = (size_t)blockIdx.x * blockDim.x + threadIdx.x;
    if (idx >= (size_t)M * Fp) return;
    int    j = (int)(idx % Fp);
    size_t i = idx / Fp;
    float v = 0.f;
    if (j < F) {
        float x = X[i * (size_t)F + j];
        v = (x - mean[j]) * rstd[j] * g[j] + be[j];
        v = v > 0.f ? v : 0.f;
    }
    dst[i * (size_t)ldo + colOff + j] = f2bf(v);
}

// BN + ReLU, f32 output (used for the proj block feeding the fc head).
__global__ void k_bn_relu_f32(const float* __restrict__ X,
                              const float* __restrict__ mean, const float* __restrict__ rstd,
                              const float* __restrict__ g, const float* __restrict__ be,
                              float* __restrict__ dst, int M, int F)
{
    size_t idx = (size_t)blockIdx.x * blockDim.x + threadIdx.x;
    if (idx >= (size_t)M * F) return;
    int j = (int)(idx % F);
    float v = (X[idx] - mean[j]) * rstd[j] * g[j] + be[j];
    dst[idx] = v > 0.f ? v : 0.f;
}

// ---------------------------------------------------------------------------
// GAT source/target attention terms.  hcat layout: [MBN, 128] head-concat
// (hcat[m, h*32+o]).  s/t layout: [H, MBN].
// ---------------------------------------------------------------------------
__global__ void k_st(const float* __restrict__ hcat,
                     const float* __restrict__ a1, const float* __restrict__ a2,
                     float* __restrict__ s, float* __restrict__ t, int M, int HM)
{
    int idx = blockIdx.x * blockDim.x + threadIdx.x;
    if (idx >= HM) return;
    int h = idx / M;
    int m = idx - h * M;
    const float* hv = hcat + (size_t)m * 128 + h * 32;
    const float* A1 = a1 + h * 32;
    const float* A2 = a2 + h * 32;
    float sv = 0.f, tv = 0.f;
#pragma unroll
    for (int o = 0; o < 32; ++o) { float x = hv[o]; sv += x * A1[o]; tv += x * A2[o]; }
    s[idx] = sv;
    t[idx] = tv;
}

// ---------------------------------------------------------------------------
// Fused masked-softmax attention row + aggregation + ELU + bf16 pack.
// Grid: (N, H, B), block 128.  Avoids materializing att[B,H,N,N] (164 MB).
// ---------------------------------------------------------------------------
__global__ void __launch_bounds__(128)
k_attn(const float* __restrict__ s, const float* __restrict__ t,
       const float* __restrict__ hcat, const int* __restrict__ Adj,
       bf16* __restrict__ xout, int B, int N, int H, int Fo, int ldo)
{
    const int n = blockIdx.x, h = blockIdx.y, b = blockIdx.z;
    const int tid = threadIdx.x;
    __shared__ float att[128];
    __shared__ float red[128];

    const size_t stBase = (size_t)h * B * N + (size_t)b * N;
    const float  sn = s[stBase + n];

    float e = -1e9f;
    if (tid < N) {
        int a = Adj[((size_t)b * N + n) * N + tid];
        float ev = sn + t[stBase + tid];
        ev = ev > 0.f ? ev : 0.2f * ev;          // LeakyReLU(0.2)
        e = (a > 0) ? ev : -1e9f;
    }
    red[tid] = e;
    __syncthreads();
    for (int st = 64; st > 0; st >>= 1) {
        if (tid < st) red[tid] = fmaxf(red[tid], red[tid + st]);
        __syncthreads();
    }
    const float mx = red[0];
    __syncthreads();

    const float ex = (tid < N) ? __expf(e - mx) : 0.f;
    att[tid] = ex;
    red[tid] = ex;
    __syncthreads();
    for (int st = 64; st > 0; st >>= 1) {
        if (tid < st) red[tid] += red[tid + st];
        __syncthreads();
    }
    const float inv = 1.f / red[0];

    if (tid < Fo) {
        // hcat[(b*N + m), h*32 + tid], stride 128 per neighbor m
        const float* hcol = hcat + ((size_t)b * N) * 128 + h * Fo + tid;
        float acc = 0.f;
        for (int m = 0; m < N; ++m) acc += att[m] * hcol[(size_t)m * 128];
        acc *= inv;
        float v = acc > 0.f ? acc : __expf(acc) - 1.f;   // ELU
        xout[((size_t)b * N + n) * ldo + h * Fo + tid] = f2bf(v);
    }
}

// ---------------------------------------------------------------------------
// fc head: p[b] = sigmoid(x[b,:128].W + b)
// ---------------------------------------------------------------------------
__global__ void k_fc(const float* __restrict__ xp, const float* __restrict__ W,
                     const float* __restrict__ bfc, float* __restrict__ p, int Bn)
{
    int b = blockIdx.x * blockDim.x + threadIdx.x;
    if (b >= Bn) return;
    float acc = bfc[0];
#pragma unroll 4
    for (int j = 0; j < 128; ++j) acc += xp[(size_t)b * 128 + j] * W[j];
    p[b] = 1.f / (1.f + __expf(-acc));
}

// BCE loss reduction (single block).
__global__ void k_loss(const float* __restrict__ p, const float* __restrict__ label,
                       float* __restrict__ out, int Bn)
{
    __shared__ float sh[256];
    float acc = 0.f;
    for (int i = threadIdx.x; i < Bn; i += 256) {
        float pc = fminf(fmaxf(p[i], 1e-7f), 1.f - 1e-7f);
        acc += label[i] * __logf(pc) + (1.f - label[i]) * __logf(1.f - pc);
    }
    sh[threadIdx.x] = acc;
    __syncthreads();
    for (int st = 128; st > 0; st >>= 1) {
        if (threadIdx.x < st) sh[threadIdx.x] += sh[threadIdx.x + st];
        __syncthreads();
    }
    if (threadIdx.x == 0) out[0] = -sh[0] / Bn;
}

// ---------------------------------------------------------------------------
// Host-side orchestration
// ---------------------------------------------------------------------------
extern "C" void kernel_launch(void* const* d_in, const int* in_sizes, int n_in,
                              void* d_out, int out_size, void* d_ws, size_t ws_size,
                              hipStream_t stream)
{
    (void)in_sizes; (void)n_in; (void)out_size; (void)ws_size;

    constexpr int BATCH = 1024, NN = 100, FIN = 65, NH = 4, FO = 32, NL = 5;
    constexpr int MBN = BATCH * NN;      // 102400
    constexpr int HF  = NH * FO;         // 128
    constexpr int K0P = 96;              // 65 padded to 96
    constexpr int GK  = HF * NN;         // 12800 (gproj in-features, already %32==0)

    // Data inputs (dict insertion order).
    const float* f_in[5] = { (const float*)d_in[0], (const float*)d_in[1],
                             (const float*)d_in[2], (const float*)d_in[3],
                             (const float*)d_in[4] };
    const float* Xf    = (const float*)d_in[5];
    const int*   Adj   = (const int*)  d_in[6];
    const float* label = (const float*)d_in[7];

    // params pytree flatten (jax sorts dict keys): fc, fp, gat, gproj, proj.
    auto P = [&](int i) -> const float* { return (const float*)d_in[i]; };
    const float* fcW = P(8);
    const float* fcb = P(9);
    auto fpP    = [&](int br, int blk, int k) { return P(10 + br * 12 + blk * 4 + k); };
    auto gatW   = [&](int l) { return P(70 + l * 3 + 0); };
    auto gatA1  = [&](int l) { return P(70 + l * 3 + 1); };
    auto gatA2  = [&](int l) { return P(70 + l * 3 + 2); };
    auto gprojP = [&](int blk, int k) { return P(85 + blk * 4 + k); };
    auto projP  = [&](int k) { return P(93 + k); };

    // Workspace bump allocator (256B alignment).
    char*  wsp = (char*)d_ws;
    size_t off = 0;
    auto alloc = [&](size_t bytes) -> void* {
        void* p = wsp + off;
        off += (bytes + 255) & ~(size_t)255;
        return p;
    };

    bf16*  Xbf   = (bf16*) alloc((size_t)MBN * K0P * 2);      // GAT layer-0 input
    bf16*  xA    = (bf16*) alloc((size_t)MBN * HF  * 2);      // GAT ping
    bf16*  xB    = (bf16*) alloc((size_t)MBN * HF  * 2);      // GAT pong
    float* hbuf  = (float*)alloc((size_t)MBN * HF * 4);       // head-concat features
    float* sbuf  = (float*)alloc((size_t)NH * MBN * 4);
    float* tbuf  = (float*)alloc((size_t)NH * MBN * 4);
    bf16*  gatWt = (bf16*) alloc((size_t)HF * HF * 2);        // [128, Kp<=128] transposed head weights
    bf16*  Wt    = (bf16*) alloc((size_t)1024 * GK * 2);      // largest transposed weight
    float* cbuf  = (float*)alloc((size_t)BATCH * 1024 * 4);   // GEMM f32 output
    float* meanb = (float*)alloc(1024 * 4);
    float* rstdb = (float*)alloc(1024 * 4);
    bf16*  actA  = (bf16*) alloc((size_t)BATCH * 4864 * 2);   // branch / gproj activations
    bf16*  actB  = (bf16*) alloc((size_t)BATCH * 512  * 2);
    bf16*  comb  = (bf16*) alloc((size_t)BATCH * 768  * 2);   // concat [g, fp0..fp4]
    float* xproj = (float*)alloc((size_t)BATCH * 128  * 4);

    auto cdiv = [](size_t a, size_t b) { return (unsigned)((a + b - 1) / b); };

    auto gemm = [&](const bf16* A, const bf16* Bt, float* C, const float* bias,
                    int M, int N, int Kp) {
        dim3 g(N / 64, M / 128);
        k_gemm_bf16<<<g, 256, 0, stream>>>(A, Bt, C, bias, M, N, Kp);
    };

    // One MLP _block: linear(+bias) -> BN(train) -> ReLU -> bf16 (padded / sliced).
    auto mlp_block = [&](const bf16* Abf, int M, int K, int Kp, int Nout,
                         const float* W, const float* b, const float* g_, const float* be,
                         bf16* dst, int ldo, int colOff, int Fp) {
        k_cast_transpose<<<cdiv((size_t)Nout * Kp, 256), 256, 0, stream>>>(W, Wt, K, Nout, Kp);
        gemm(Abf, Wt, cbuf, b, M, Nout, Kp);
        k_bn_stats<<<Nout, 256, 0, stream>>>(cbuf, M, Nout, meanb, rstdb);
        k_bn_relu_bf16<<<cdiv((size_t)M * Fp, 256), 256, 0, stream>>>(
            cbuf, meanb, rstdb, g_, be, dst, M, Nout, ldo, colOff, Fp);
    };

    // ---------------- GAT stack (5 layers) ----------------
    k_cast_pad<<<cdiv((size_t)MBN * K0P, 256), 256, 0, stream>>>(Xf, Xbf, MBN, FIN, K0P);

    const bf16* xin = Xbf;
    int Kp = K0P;
    for (int l = 0; l < NL; ++l) {
        const int fi = (l == 0) ? FIN : HF;
        // Concatenate transposed per-head weights into Bt_all [128, Kp]:
        // row (h*32+o) holds W_h[:, o] -> one fused GEMM over all 4 heads.
        for (int h = 0; h < NH; ++h)
            k_cast_transpose<<<cdiv((size_t)FO * Kp, 256), 256, 0, stream>>>(
                gatW(l) + (size_t)h * fi * FO, gatWt + (size_t)h * FO * Kp, fi, FO, Kp);
        gemm(xin, gatWt, hbuf, nullptr, MBN, HF, Kp);   // hbuf[m, h*32+o]

        const int HM = NH * MBN;
        k_st<<<cdiv(HM, 256), 256, 0, stream>>>(hbuf, gatA1(l), gatA2(l),
                                                sbuf, tbuf, MBN, HM);

        bf16* xout = (l & 1) ? xB : xA;
        dim3 ga(NN, NH, BATCH);
        k_attn<<<ga, 128, 0, stream>>>(sbuf, tbuf, hbuf, Adj, xout,
                                       BATCH, NN, NH, FO, HF);
        xin = xout;
        Kp  = HF;
    }
    // xin is [B*N, 128] bf16 == g reshaped to [B, 12800] (same memory layout).
    const bf16* gbf = xin;

    // ---------------- gproj: 12800 -> 1024 -> 128 ----------------
    mlp_block(gbf, BATCH, GK, GK, 1024,
              gprojP(0, 0), gprojP(0, 1), gprojP(0, 2), gprojP(0, 3),
              actA, 1024, 0, 1024);
    mlp_block(actA, BATCH, 1024, 1024, 128,
              gprojP(1, 0), gprojP(1, 1), gprojP(1, 2), gprojP(1, 3),
              comb, 768, 0, 128);

    // ---------------- 5 feature branches: d -> 512 -> 256 -> 128 ----------------
    const int bd[5]  = {1024, 881, 1024, 780, 4860};
    const int bdp[5] = {1024, 896, 1024, 800, 4864};
    for (int i = 0; i < 5; ++i) {
        k_cast_pad<<<cdiv((size_t)BATCH * bdp[i], 256), 256, 0, stream>>>(
            f_in[i], actA, BATCH, bd[i], bdp[i]);
        mlp_block(actA, BATCH, bd[i], bdp[i], 512,
                  fpP(i, 0, 0), fpP(i, 0, 1), fpP(i, 0, 2), fpP(i, 0, 3),
                  actB, 512, 0, 512);
        mlp_block(actB, BATCH, 512, 512, 256,
                  fpP(i, 1, 0), fpP(i, 1, 1), fpP(i, 1, 2), fpP(i, 1, 3),
                  actA, 256, 0, 256);
        mlp_block(actA, BATCH, 256, 256, 128,
                  fpP(i, 2, 0), fpP(i, 2, 1), fpP(i, 2, 2), fpP(i, 2, 3),
                  comb, 768, 128 * (i + 1), 128);
    }

    // ---------------- proj block: 768 -> 128 (f32 out) ----------------
    k_cast_transpose<<<cdiv((size_t)128 * 768, 256), 256, 0, stream>>>(
        projP(0), Wt, 768, 128, 768);
    gemm(comb, Wt, cbuf, projP(1), BATCH, 128, 768);
    k_bn_stats<<<128, 256, 0, stream>>>(cbuf, BATCH, 128, meanb, rstdb);
    k_bn_relu_f32<<<cdiv((size_t)BATCH * 128, 256), 256, 0, stream>>>(
        cbuf, meanb, rstdb, projP(2), projP(3), xproj, BATCH, 128);

    // ---------------- fc head + sigmoid + BCE loss ----------------
    float* out = (float*)d_out;                 // [0..1023] = p, [1024] = loss
    k_fc<<<cdiv(BATCH, 256), 256, 0, stream>>>(xproj, fcW, fcb, out, BATCH);
    k_loss<<<1, 256, 0, stream>>>(out, label, out + BATCH, BATCH);
}
